// GCNBaseline_45535243272660
// MI455X (gfx1250) — matrix operand order repaired and checked by reading, not compiled, not run on hardware
//
#include <hip/hip_runtime.h>
#include <math.h>

// ---- types -----------------------------------------------------------------
typedef _Float16 h8  __attribute__((ext_vector_type(8)));
typedef _Float16 h16 __attribute__((ext_vector_type(16)));
typedef float    f8  __attribute__((ext_vector_type(8)));
typedef float    f4  __attribute__((ext_vector_type(4)));

#define HID 128
#define NHEAD 4
#define HC 32

__device__ __forceinline__ float silu_f(float y) { return y / (1.f + __expf(-y)); }

__device__ __forceinline__ void atomicMaxFloat(float* addr, float v) {
    // order-preserving int trick (handles mixed signs, init must be -inf)
    if (v >= 0.f) atomicMax((int*)addr, __float_as_int(v));
    else          atomicMin((unsigned int*)addr, __float_as_uint(v));
}

// ---- generic fill ----------------------------------------------------------
__global__ void fill_kernel(float* p, float v, int n) {
    int i = blockIdx.x * blockDim.x + threadIdx.x;
    if (i < n) p[i] = v;
}

// ---- weight transpose: Wg[l][k][n] -> Wt16[l][n][k] (f16) ------------------
__global__ void wtrans_kernel(const float* __restrict__ Wg, _Float16* __restrict__ Wt) {
    int idx = blockIdx.x * blockDim.x + threadIdx.x;
    if (idx >= 4 * 128 * 128) return;
    int l = idx >> 14, rem = idx & 16383, n = rem >> 7, k = rem & 127;
    Wt[idx] = (_Float16)Wg[l * 16384 + k * 128 + n];
}

// ---- input projection: concat(x,pos) @ W_in + b_in -> out[N,128] ----------
__global__ void input_proj_kernel(const float* __restrict__ x, const float* __restrict__ pos,
                                  const float* __restrict__ Win, const float* __restrict__ bin,
                                  float* __restrict__ out, int N) {
    int idx = blockIdx.x * blockDim.x + threadIdx.x;
    if (idx >= N * HID) return;
    int n = idx >> 7, c = idx & 127;
    float acc = bin[c];
    const float* xr = x + n * 11;
    const float* pr = pos + n * 3;
#pragma unroll
    for (int k = 0; k < 11; ++k) acc += xr[k] * Win[k * 128 + c];
#pragma unroll
    for (int k = 0; k < 3; ++k) acc += pr[k] * Win[(11 + k) * 128 + c];
    out[idx] = acc;
}

// ---- BN stats: per-channel sum / sumsq over rows (optional bias added) -----
__global__ void bn_stats_kernel(const float* __restrict__ x, const float* __restrict__ bias,
                                float* __restrict__ sums, float* __restrict__ sumsq, int N) {
    int c = threadIdx.x & 127;
    int half = threadIdx.x >> 7; // 0/1
    float bb = bias ? bias[c] : 0.f;
    float s = 0.f, q = 0.f;
    for (int n = blockIdx.x * 2 + half; n < N; n += gridDim.x * 2) {
        float v = x[(size_t)n * HID + c] + bb;
        s += v; q += v * v;
    }
    __shared__ float sh[256], sh2[256];
    sh[threadIdx.x] = s; sh2[threadIdx.x] = q;
    __syncthreads();
    if (threadIdx.x < 128) {
        atomicAdd(&sums[c],  sh[c] + sh[c + 128]);
        atomicAdd(&sumsq[c], sh2[c] + sh2[c + 128]);
    }
}

__global__ void bn_finalize_kernel(const float* sums, const float* sumsq,
                                   float* mean, float* rstd, int N) {
    int c = threadIdx.x;
    float m = sums[c] / (float)N;
    float v = sumsq[c] / (float)N - m * m;
    mean[c] = m;
    rstd[c] = rsqrtf(v + 1e-5f);
}

// ---- BN apply + residual + SiLU; writes f32 h and f16 shadow ---------------
__global__ void bn_silu_kernel(const float* __restrict__ x, const float* __restrict__ bias,
                               const float* __restrict__ res,
                               const float* __restrict__ mean, const float* __restrict__ rstd,
                               const float* __restrict__ g, const float* __restrict__ b,
                               float* __restrict__ hout, _Float16* __restrict__ hout16, int total) {
    int idx = blockIdx.x * blockDim.x + threadIdx.x;
    if (idx >= total) return;
    int c = idx & 127;
    float xx = x[idx] + (bias ? bias[c] : 0.f);
    float y = (xx - mean[c]) * rstd[c] * g[c] + b[c];
    if (res) y += res[idx];
    y = silu_f(y);
    hout[idx] = y;
    hout16[idx] = (_Float16)y;
}

// ---- WMMA GEMM: z[N,128] = h16[N,128] @ Wt16^T (Wt16 is [n][k] f16) --------
// One 16x16 tile per wave; 8 waves/block cover all 128 output cols of a 16-row
// strip.  N must be a multiple of 16 (100000 = 6250*16).  EXEC is all-ones.
__global__ void __launch_bounds__(256) gemm_wmma_kernel(const _Float16* __restrict__ A,
                                                        const _Float16* __restrict__ Bt,
                                                        float* __restrict__ C) {
    const int lane = threadIdx.x & 31;
    const int wave = threadIdx.x >> 5;          // n-tile 0..7
    const int m0 = blockIdx.x * 16;
    const int n0 = wave * 16;
    const int r  = lane & 15;
    const int hi = lane >> 4;                   // lane half
    const _Float16* arow = A  + (size_t)(m0 + r) * HID;
    const _Float16* brow = Bt + (size_t)(n0 + r) * HID;
    f8 acc = {};
#pragma unroll
    for (int kk = 0; kk < 4; ++kk) {
        const int kb = kk * 32;
        // A (16x32 f16): lane<16 holds K {kb..kb+7, kb+16..kb+23}; lane>=16 +8
        h8 a0 = *(const h8*)(arow + kb + hi * 8);
        h8 a1 = *(const h8*)(arow + kb + 16 + hi * 8);
        // B (32x16 f16): lane<16 = col n0+r with K {kb..kb+7, kb+16..kb+23}; lane>=16 +8
        h8 b0 = *(const h8*)(brow + kb + hi * 8);
        h8 b1 = *(const h8*)(brow + kb + 16 + hi * 8);
        h16 av, bv;
#pragma unroll
        for (int i = 0; i < 8; ++i) { av[i] = a0[i]; av[i + 8] = a1[i];
                                      bv[i] = b0[i]; bv[i + 8] = b1[i]; }
        acc = __builtin_amdgcn_wmma_f32_16x16x32_f16(false, av, false, bv,
                                                     (short)0, acc, false, false);
    }
    // D layout: VGPR v -> (M = v + hi*8, N = n0 + (lane&15))
    float* crow = C + (size_t)(m0 + hi * 8) * HID + n0 + r;
#pragma unroll
    for (int v = 0; v < 8; ++v) crow[(size_t)v * HID] = acc[v];
}

// ---- per-node attention coefficients: es/ed [N,4] --------------------------
__global__ void attn_coef_kernel(const float* __restrict__ z,
                                 const float* __restrict__ asrc, const float* __restrict__ adst,
                                 float* __restrict__ es, float* __restrict__ ed, int N) {
    int idx = blockIdx.x * blockDim.x + threadIdx.x;
    if (idx >= N * NHEAD) return;
    int n = idx >> 2, hh = idx & 3;
    const float* zr = z + (size_t)n * HID + hh * HC;
    const float* as = asrc + hh * HC;
    const float* ad = adst + hh * HC;
    float a = 0.f, b = 0.f;
#pragma unroll
    for (int c = 0; c < HC; ++c) { float v = zr[c]; a += v * as[c]; b += v * ad[c]; }
    es[idx] = a; ed[idx] = b;
}

// ---- edge pass 1: segment max of leaky_relu(es[src]+ed[dst]) ---------------
__global__ void edge_max_kernel(const int* __restrict__ ei, int E, int N,
                                const float* __restrict__ es, const float* __restrict__ ed,
                                float* __restrict__ m) {
    int e = blockIdx.x * blockDim.x + threadIdx.x;
    if (e >= E + N) return;
    int s, d;
    if (e < E) { s = ei[e]; d = ei[E + e]; } else { s = d = e - E; }
    f4 a = *(const f4*)(es + s * 4);
    f4 b = *(const f4*)(ed + d * 4);
#pragma unroll
    for (int hh = 0; hh < 4; ++hh) {
        float v = a[hh] + b[hh];
        v = v > 0.f ? v : 0.2f * v;
        atomicMaxFloat(m + d * 4 + hh, v);
    }
}

// ---- edge pass 2: denom = segment_sum(exp(e - m[dst])) ---------------------
__global__ void edge_expsum_kernel(const int* __restrict__ ei, int E, int N,
                                   const float* __restrict__ es, const float* __restrict__ ed,
                                   const float* __restrict__ m, float* __restrict__ den) {
    int e = blockIdx.x * blockDim.x + threadIdx.x;
    if (e >= E + N) return;
    int s, d;
    if (e < E) { s = ei[e]; d = ei[E + e]; } else { s = d = e - E; }
    f4 a = *(const f4*)(es + s * 4);
    f4 b = *(const f4*)(ed + d * 4);
    f4 mm = *(const f4*)(m + d * 4);
#pragma unroll
    for (int hh = 0; hh < 4; ++hh) {
        float v = a[hh] + b[hh];
        v = v > 0.f ? v : 0.2f * v;
        atomicAdd(den + d * 4 + hh, __expf(v - mm[hh]));
    }
}

// ---- edge pass 3: h2[dst] += z[src] * a  (wave per edge, 4 ch/lane) --------
__global__ void __launch_bounds__(256) edge_agg_kernel(const int* __restrict__ ei, int E, int N,
                                const float* __restrict__ es, const float* __restrict__ ed,
                                const float* __restrict__ m, const float* __restrict__ den,
                                const float* __restrict__ z, float* __restrict__ h2) {
    int e = blockIdx.x * 8 + (threadIdx.x >> 5);
    if (e >= E + N) return;
    int lane = threadIdx.x & 31;
    int s, d;
    if (e < E) { s = ei[e]; d = ei[E + e]; } else { s = d = e - E; }
    int hh = lane >> 3;  // head for channels lane*4..lane*4+3
    float v = es[s * 4 + hh] + ed[d * 4 + hh];
    v = v > 0.f ? v : 0.2f * v;
    float a = __expf(v - m[d * 4 + hh]) / den[d * 4 + hh];
    int c0 = lane * 4;
    f4 zv = *(const f4*)(z + (size_t)s * HID + c0);
    float* out = h2 + (size_t)d * HID + c0;
#pragma unroll
    for (int j = 0; j < 4; ++j) atomicAdd(out + j, zv[j] * a);
}

// ---- pooling: s[n] = h[n] . W_att + b_att (wave per node) ------------------
__global__ void score_kernel(const float* __restrict__ h, const float* __restrict__ Watt,
                             const float* __restrict__ batt, float* __restrict__ s, int N) {
    int node = blockIdx.x * 8 + (threadIdx.x >> 5);
    if (node >= N) return;
    int lane = threadIdx.x & 31;
    int c0 = lane * 4;
    float acc = 0.f;
#pragma unroll
    for (int j = 0; j < 4; ++j) acc += h[(size_t)node * HID + c0 + j] * Watt[c0 + j];
    for (int off = 16; off; off >>= 1) acc += __shfl_xor(acc, off, 32);
    if (lane == 0) s[node] = acc + batt[0];
}

__global__ void reduce_max_kernel(const float* __restrict__ s, int N, float* gmax) {
    float m = -INFINITY;
    for (int i = blockIdx.x * blockDim.x + threadIdx.x; i < N; i += gridDim.x * blockDim.x)
        m = fmaxf(m, s[i]);
    for (int off = 16; off; off >>= 1) m = fmaxf(m, __shfl_xor(m, off, 32));
    if ((threadIdx.x & 31) == 0) atomicMaxFloat(gmax, m);
}

__global__ void expsum_kernel(const float* __restrict__ s, const float* __restrict__ gmax,
                              float* __restrict__ t, float* gsum, int N) {
    float M = *gmax, acc = 0.f;
    for (int i = blockIdx.x * blockDim.x + threadIdx.x; i < N; i += gridDim.x * blockDim.x) {
        float e = __expf(s[i] - M);
        t[i] = e; acc += e;
    }
    for (int off = 16; off; off >>= 1) acc += __shfl_xor(acc, off, 32);
    if ((threadIdx.x & 31) == 0) atomicAdd(gsum, acc);
}

__global__ void w_segmax_kernel(const float* __restrict__ t, const float* __restrict__ gsum,
                                const int* __restrict__ batch, float* __restrict__ w,
                                float* __restrict__ mw, int N) {
    int n = blockIdx.x * blockDim.x + threadIdx.x;
    if (n >= N) return;
    float wv = t[n] / (*gsum);
    w[n] = wv;
    atomicMaxFloat(&mw[batch[n]], wv);
}

__global__ void pw_segsum_kernel(const float* __restrict__ w, const float* __restrict__ mw,
                                 const int* __restrict__ batch, float* __restrict__ pw,
                                 float* __restrict__ sden, int N) {
    int n = blockIdx.x * blockDim.x + threadIdx.x;
    if (n >= N) return;
    int b = batch[n];
    float p = __expf(w[n] - mw[b]);
    pw[n] = p;
    atomicAdd(&sden[b], p);
}

__global__ void wsum_kernel(const float* __restrict__ h, const float* __restrict__ pw,
                            const float* __restrict__ sden, const int* __restrict__ batch,
                            float* __restrict__ hg, int N) {
    int node = blockIdx.x * 8 + (threadIdx.x >> 5);
    if (node >= N) return;
    int lane = threadIdx.x & 31;
    int b = batch[node];
    float sw = pw[node] / sden[b];
    int c0 = lane * 4;
#pragma unroll
    for (int j = 0; j < 4; ++j)
        atomicAdd(&hg[b * HID + c0 + j], h[(size_t)node * HID + c0 + j] * sw);
}

// ---- output MLP: one block (128 thr) per graph -----------------------------
__global__ void mlp_kernel(const float* __restrict__ hg,
                           const float* __restrict__ W1, const float* __restrict__ b1,
                           const float* __restrict__ W2, const float* __restrict__ b2,
                           const float* __restrict__ W3, const float* __restrict__ b3,
                           float* __restrict__ pred) {
    __shared__ float sh[128], s1[128], s2[64];
    int g = blockIdx.x, t = threadIdx.x;
    sh[t] = hg[g * 128 + t];
    __syncthreads();
    float acc = b1[t];
    for (int k = 0; k < 128; ++k) acc += sh[k] * W1[k * 128 + t];
    s1[t] = silu_f(acc);
    __syncthreads();
    if (t < 64) {
        float a2 = b2[t];
        for (int k = 0; k < 128; ++k) a2 += s1[k] * W2[k * 64 + t];
        s2[t] = silu_f(a2);
    }
    __syncthreads();
    if (t == 0) {
        float a3 = b3[0];
        for (int k = 0; k < 64; ++k) a3 += s2[k] * W3[k];
        pred[g] = a3;
    }
}

// ---- host orchestration ----------------------------------------------------
extern "C" void kernel_launch(void* const* d_in, const int* in_sizes, int n_in,
                              void* d_out, int out_size, void* d_ws, size_t ws_size,
                              hipStream_t stream) {
    const float* x    = (const float*)d_in[0];
    const float* pos  = (const float*)d_in[1];
    const int*   ei   = (const int*)d_in[2];
    const int*   batch= (const int*)d_in[3];
    const float* Win  = (const float*)d_in[4];
    const float* bin  = (const float*)d_in[5];
    const float* in_g = (const float*)d_in[6];
    const float* in_b = (const float*)d_in[7];
    const float* Wg   = (const float*)d_in[8];
    const float* asrc = (const float*)d_in[9];
    const float* adst = (const float*)d_in[10];
    const float* bg   = (const float*)d_in[11];
    const float* ln_g = (const float*)d_in[12];
    const float* ln_b = (const float*)d_in[13];
    const float* Watt = (const float*)d_in[14];
    const float* batt = (const float*)d_in[15];
    const float* W1   = (const float*)d_in[16];
    const float* b1   = (const float*)d_in[17];
    const float* W2   = (const float*)d_in[18];
    const float* b2   = (const float*)d_in[19];
    const float* W3   = (const float*)d_in[20];
    const float* b3   = (const float*)d_in[21];

    const int N = in_sizes[3];
    const int E = in_sizes[2] / 2;
    const int G = out_size - N * HID;          // 256
    const int NE = E + N;
    const int NH_TOT = N * HID;

    float* pred = (float*)d_out;
    float* h    = (float*)d_out + G;           // final h lives directly in d_out

    // workspace carve-out (256B aligned)
    size_t off = 0;
    char* base = (char*)d_ws;
    auto carve = [&](size_t bytes) -> char* {
        char* p = base + off;
        off = (off + bytes + 255) & ~(size_t)255;
        return p;
    };
    _Float16* h16  = (_Float16*)carve((size_t)NH_TOT * 2);
    float* z       = (float*)carve((size_t)NH_TOT * 4);   // also used as hpre
    float* h2      = (float*)carve((size_t)NH_TOT * 4);
    float* es      = (float*)carve((size_t)N * 4 * 4);
    float* ed      = (float*)carve((size_t)N * 4 * 4);
    float* m       = (float*)carve((size_t)N * 4 * 4);
    float* den     = (float*)carve((size_t)N * 4 * 4);
    _Float16* Wt16 = (_Float16*)carve((size_t)4 * 128 * 128 * 2);
    float* sums    = (float*)carve(256 * 4);              // sums[128] + sumsq[128]
    float* sumsq   = sums + 128;
    float* meanv   = (float*)carve(256 * 4);              // mean[128] + rstd[128]
    float* rstd    = meanv + 128;
    float* sc      = (float*)carve((size_t)N * 4);        // pooling score s
    float* tbuf    = (float*)carve((size_t)N * 4);        // exp(s - gmax)
    float* wbuf    = (float*)carve((size_t)N * 4);        // global-softmax w
    float* pwbuf   = (float*)carve((size_t)N * 4);
    float* mw      = (float*)carve((size_t)G * 4);
    float* sden    = (float*)carve((size_t)G * 4);
    float* hg      = (float*)carve((size_t)G * HID * 4);
    float* gscal   = (float*)carve(2 * 4);                // gmax, gsum

    const int T = 256;
    auto blk = [](int n, int t) { return (n + t - 1) / t; };

    // weights -> f16 transposed (N-major) for WMMA B operand
    wtrans_kernel<<<blk(4 * 128 * 128, T), T, 0, stream>>>(Wg, Wt16);

    // input projection + BN + SiLU
    input_proj_kernel<<<blk(NH_TOT, T), T, 0, stream>>>(x, pos, Win, bin, z, N);
    fill_kernel<<<1, 256, 0, stream>>>(sums, 0.f, 256);
    bn_stats_kernel<<<512, 256, 0, stream>>>(z, nullptr, sums, sumsq, N);
    bn_finalize_kernel<<<1, 128, 0, stream>>>(sums, sumsq, meanv, rstd, N);
    bn_silu_kernel<<<blk(NH_TOT, T), T, 0, stream>>>(z, nullptr, nullptr, meanv, rstd,
                                                     in_g, in_b, h, h16, NH_TOT);

    // GAT layers
    for (int l = 0; l < 4; ++l) {
        gemm_wmma_kernel<<<N / 16, 256, 0, stream>>>(h16, Wt16 + (size_t)l * 16384, z);
        attn_coef_kernel<<<blk(N * 4, T), T, 0, stream>>>(z, asrc + l * 128, adst + l * 128,
                                                          es, ed, N);
        fill_kernel<<<blk(N * 4, T), T, 0, stream>>>(m, -INFINITY, N * 4);
        fill_kernel<<<blk(N * 4, T), T, 0, stream>>>(den, 0.f, N * 4);
        fill_kernel<<<blk(NH_TOT, T), T, 0, stream>>>(h2, 0.f, NH_TOT);
        edge_max_kernel<<<blk(NE, T), T, 0, stream>>>(ei, E, N, es, ed, m);
        edge_expsum_kernel<<<blk(NE, T), T, 0, stream>>>(ei, E, N, es, ed, m, den);
        edge_agg_kernel<<<blk(NE, 8), 256, 0, stream>>>(ei, E, N, es, ed, m, den, z, h2);
        fill_kernel<<<1, 256, 0, stream>>>(sums, 0.f, 256);
        bn_stats_kernel<<<512, 256, 0, stream>>>(h2, bg + l * 128, sums, sumsq, N);
        bn_finalize_kernel<<<1, 128, 0, stream>>>(sums, sumsq, meanv, rstd, N);
        bn_silu_kernel<<<blk(NH_TOT, T), T, 0, stream>>>(h2, bg + l * 128, /*res=*/h,
                                                         meanv, rstd, ln_g + l * 128,
                                                         ln_b + l * 128, h, h16, NH_TOT);
    }

    // attention pooling (global softmax then per-graph softmax, as in reference)
    score_kernel<<<blk(N, 8), 256, 0, stream>>>(h, Watt, batt, sc, N);
    fill_kernel<<<1, 32, 0, stream>>>(gscal, -INFINITY, 1);
    fill_kernel<<<1, 32, 0, stream>>>(gscal + 1, 0.f, 1);
    reduce_max_kernel<<<512, 256, 0, stream>>>(sc, N, gscal);
    expsum_kernel<<<512, 256, 0, stream>>>(sc, gscal, tbuf, gscal + 1, N);
    fill_kernel<<<blk(G, T), T, 0, stream>>>(mw, -INFINITY, G);
    w_segmax_kernel<<<blk(N, T), T, 0, stream>>>(tbuf, gscal + 1, batch, wbuf, mw, N);
    fill_kernel<<<blk(G, T), T, 0, stream>>>(sden, 0.f, G);
    pw_segsum_kernel<<<blk(N, T), T, 0, stream>>>(wbuf, mw, batch, pwbuf, sden, N);
    fill_kernel<<<blk(G * HID, T), T, 0, stream>>>(hg, 0.f, G * HID);
    wsum_kernel<<<blk(N, 8), 256, 0, stream>>>(h, pwbuf, sden, batch, hg, N);

    // output MLP
    mlp_kernel<<<G, 128, 0, stream>>>(hg, W1, b1, W2, b2, W3, b3, pred);
}
